// Kernel_Conv_59210419143342
// MI455X (gfx1250) — matrix operand order repaired and checked by reading, not compiled, not run on hardware
//
#include <hip/hip_runtime.h>
#include <math.h>

// ---------------- problem constants ----------------
#define GB   32          // batch
#define GC   256         // in channels
#define GO   256         // out channels
#define GH   56
#define GW   56
#define GHW  3136        // 56*56
#define TAPS 9

// ---------------- GEMM tiling ----------------
#define BM   64          // spatial rows per block  (3136 = 49*64, exact)
#define BN   128         // out channels per block  (256 = 2*128, exact)
#define KB   32          // K per WMMA step (bf16)
#define NK   72          // 9 taps * (256/32) c-steps
#define LDA  40          // As row pitch in bf16 (32 + 8 pad -> 80B rows)
#define LDB  40          // Bs row pitch in bf16

typedef __attribute__((ext_vector_type(16))) __bf16 v16bf;
typedef __attribute__((ext_vector_type(8)))  float  v8f;

// =====================================================================
// 1) x fp32 -> bf16 (8 elements / thread, exact coverage)
// =====================================================================
__global__ __launch_bounds__(256) void cvt_kernel(const float* __restrict__ x,
                                                  __bf16* __restrict__ xbf) {
  int i = blockIdx.x * 256 + threadIdx.x;           // 12544*256 == 25690112/8
  const float4* xp = (const float4*)x + (size_t)i * 2;
  float4 a = xp[0], b = xp[1];
  union { __bf16 h[8]; uint4 u; } p;
  p.h[0] = (__bf16)a.x; p.h[1] = (__bf16)a.y; p.h[2] = (__bf16)a.z; p.h[3] = (__bf16)a.w;
  p.h[4] = (__bf16)b.x; p.h[5] = (__bf16)b.y; p.h[6] = (__bf16)b.z; p.h[7] = (__bf16)b.w;
  ((uint4*)xbf)[i] = p.u;
}

// =====================================================================
// 2) fused adaptive avg+max pool 56x56 -> 5x5, one wave per (b,c)
// =====================================================================
__global__ __launch_bounds__(32) void pool_kernel(const float* __restrict__ x,
                                                  float* __restrict__ avgp,
                                                  float* __restrict__ mxp) {
  const int bc   = blockIdx.x;                  // b*256 + c
  const int lane = threadIdx.x;
  if (lane >= 25) return;
  const float* xp = x + (size_t)bc * GHW;
  const int i = lane / 5, j = lane % 5;
  const int r0 = (i * GH) / 5,       r1 = ((i + 1) * GH + 4) / 5;
  const int c0 = (j * GW) / 5,       c1 = ((j + 1) * GW + 4) / 5;
  float s = 0.0f, m = -3.402823466e+38f;
  for (int r = r0; r < r1; ++r) {
    const float* row = xp + r * GW;
    for (int c = c0; c < c1; ++c) {
      float v = row[c];
      s += v;
      m = fmaxf(m, v);
    }
  }
  float inv = 1.0f / (float)((r1 - r0) * (c1 - c0));
  avgp[(size_t)bc * 25 + lane] = s * inv;
  mxp [(size_t)bc * 25 + lane] = m;
}

// =====================================================================
// 3) depthwise 3x3 (VALID) on avg pool + bias + relu -> (B,256,3,3)
// =====================================================================
__global__ __launch_bounds__(256) void dw_relu_kernel(const float* __restrict__ avgp,
                                                      const float* __restrict__ w1,
                                                      const float* __restrict__ b1,
                                                      float* __restrict__ relup) {
  int idx = blockIdx.x * 256 + threadIdx.x;     // 32 blocks * 256 == 8192 (b,ch)
  if (idx >= GB * GC) return;
  int ch = idx & (GC - 1);
  const float* a = avgp + (size_t)idx * 25;
  const float* w = w1 + ch * 9;
  float bias = b1[ch];
#pragma unroll
  for (int i = 0; i < 3; ++i)
#pragma unroll
    for (int j = 0; j < 3; ++j) {
      float z = bias;
#pragma unroll
      for (int di = 0; di < 3; ++di)
#pragma unroll
        for (int dj = 0; dj < 3; ++dj)
          z += w[di * 3 + dj] * a[(i + di) * 5 + (j + dj)];
      relup[(size_t)idx * 9 + i * 3 + j] = fmaxf(z, 0.0f);
    }
}

// =====================================================================
// 4) dynamic weight generation:
//    W[b,tap,o,c] = (w2[oc]*relu[b,o,tap] + b2[oc]) * sigmoid(att)  (bf16)
//    block = (b,o), thread = c; layout [b][tap][o][c] (c contiguous)
// =====================================================================
__global__ __launch_bounds__(256) void wgen_kernel(const float* __restrict__ mxp,
                                                   const float* __restrict__ relup,
                                                   const float* __restrict__ w2,
                                                   const float* __restrict__ b2,
                                                   const float* __restrict__ w_att,
                                                   const float* __restrict__ b_att,
                                                   __bf16* __restrict__ wdyn) {
  const int bo = blockIdx.x;                    // b*256 + o
  const int b  = bo >> 8;
  const int o  = bo & 255;
  const int c  = threadIdx.x;
  __shared__ float ms[25], rs[9];
  if (c < 25) ms[c] = mxp[(size_t)bo * 25 + c];
  if (c < 9)  rs[c] = relup[(size_t)bo * 9 + c];
  __syncthreads();
  const int oc = o * GC + c;
  float wa[9];
#pragma unroll
  for (int t = 0; t < 9; ++t) wa[t] = w_att[(size_t)oc * 9 + t];
  const float bav = b_att[oc], w2v = w2[oc], b2v = b2[oc];
#pragma unroll
  for (int i = 0; i < 3; ++i)
#pragma unroll
    for (int j = 0; j < 3; ++j) {
      float z = bav;
#pragma unroll
      for (int di = 0; di < 3; ++di)
#pragma unroll
        for (int dj = 0; dj < 3; ++dj)
          z += wa[di * 3 + dj] * ms[(i + di) * 5 + (j + dj)];
      float att = 1.0f / (1.0f + __expf(-z));
      float wv  = w2v * rs[i * 3 + j] + b2v;
      int tap = i * 3 + j;
      wdyn[(((size_t)(b * TAPS + tap) * GO + o) * GC) + c] = (__bf16)(wv * att);
    }
}

// =====================================================================
// 5) main dynamic conv as implicit-im2col GEMM, bf16 WMMA, LDS ping-pong.
//    B tile: GLOBAL_LOAD_ASYNC_TO_LDS_B128 (ASYNCcnt) straight into LDS.
//    A tile: predicated register-staged gather (im2col with zero padding).
//    grid: (49 m-tiles, 2 n-tiles, 32 batches), 256 threads = 8 waves
// =====================================================================
__global__ __launch_bounds__(256) void dynconv_wmma_kernel(
    const __bf16* __restrict__ xbf,
    const __bf16* __restrict__ wdyn,
    float* __restrict__ out) {
  __shared__ __bf16 As[2][BM * LDA];            // 2 * 5120 B
  __shared__ __bf16 Bs[2][BN * LDB];            // 2 * 10240 B

  const int tid  = threadIdx.x;
  const int b    = blockIdx.z;
  const int m0   = blockIdx.x * BM;
  const int n0   = blockIdx.y * BN;

  const int lane = tid & 31;
  const int wave = tid >> 5;
  const int wm   = (wave & 1) * 32;             // wave M offset within tile
  const int wn   = (wave >> 1) * 32;            // wave N offset within tile

  // ---- A fill mapping: thread owns 8 consecutive channels at one m ----
  const int am  = tid & 63;                     // m within tile
  const int ag  = tid >> 6;                     // channel group (0..3)*8
  const int ayr = (m0 + am) / GW;
  const int axr = (m0 + am) % GW;

  // ---- B fill mapping: thread owns two 16B chunks ----
  const int bn_ = tid >> 2;                     // row n (0..63), +64 for 2nd
  const int bc_ = (tid & 3) * 8;                // c offset within 32

  v8f acc00 = {0,0,0,0,0,0,0,0};
  v8f acc01 = {0,0,0,0,0,0,0,0};
  v8f acc10 = {0,0,0,0,0,0,0,0};
  v8f acc11 = {0,0,0,0,0,0,0,0};

  auto loadA = [&](int tap, int c0) -> uint4 {
    const int yy = ayr + tap / 3 - 1;
    const int xx = axr + tap % 3 - 1;
    union { __bf16 h[8]; uint4 u; } r;
    if ((unsigned)yy < (unsigned)GH && (unsigned)xx < (unsigned)GW) {
      const __bf16* g = xbf + ((size_t)(b * GC + c0 + ag * 8)) * GHW + yy * GW + xx;
#pragma unroll
      for (int j = 0; j < 8; ++j) r.h[j] = g[(size_t)j * GHW];
    } else {
#pragma unroll
      for (int j = 0; j < 8; ++j) r.h[j] = (__bf16)0.0f;
    }
    return r.u;
  };

  // B tile global base for (tap, c-step) of this thread's two chunks
  auto bptr = [&](int tap, int c0) -> const __bf16* {
    return wdyn + ((size_t)((b * TAPS + tap) * GO + n0)) * GC + c0 + bc_;
  };

  // async global -> LDS copy of this thread's two B chunks (ASYNCcnt)
  auto asyncB = [&](int buf, int tap, int c0) {
    const __bf16* g  = bptr(tap, c0);
    const void*   g0 = (const void*)(g + (size_t)bn_ * GC);
    const void*   g1 = (const void*)(g + (size_t)(bn_ + 64) * GC);
    unsigned l0 = (unsigned)(unsigned long long)&Bs[buf][bn_ * LDB + bc_];
    unsigned l1 = (unsigned)(unsigned long long)&Bs[buf][(bn_ + 64) * LDB + bc_];
    asm volatile("global_load_async_to_lds_b128 %0, %1, off"
                 :: "v"(l0), "v"(g0) : "memory");
    asm volatile("global_load_async_to_lds_b128 %0, %1, off"
                 :: "v"(l1), "v"(g1) : "memory");
  };

  auto compute = [&](int buf) {
    // A frags (ISA 16-bit A 16x32 layout): lanes 0-15 K{0..7,16..23}, 16-31 K{8..15,24..31}
    union { uint4 u[2]; v16bf v; } a0, a1, b0, b1;
    const int arow = wm + (lane & 15);
    const int kb   = (lane >> 4) * 8;
    const __bf16* ap0 = &As[buf][arow * LDA + kb];
    a0.u[0] = *(const uint4*)ap0;
    a0.u[1] = *(const uint4*)(ap0 + 16);
    const __bf16* ap1 = &As[buf][(arow + 16) * LDA + kb];
    a1.u[0] = *(const uint4*)ap1;
    a1.u[1] = *(const uint4*)(ap1 + 16);
    // B frags: lanes 0-15 K{0..7,16..23} at col=lane, lanes 16-31 K{8..15,24..31}
    const int brow = wn + (lane & 15);
    const int ko   = (lane >> 4) * 8;
    const __bf16* bp0 = &Bs[buf][brow * LDB + ko];
    b0.u[0] = *(const uint4*)bp0;
    b0.u[1] = *(const uint4*)(bp0 + 16);
    const __bf16* bp1 = &Bs[buf][(brow + 16) * LDB + ko];
    b1.u[0] = *(const uint4*)bp1;
    b1.u[1] = *(const uint4*)(bp1 + 16);

    acc00 = __builtin_amdgcn_wmma_f32_16x16x32_bf16(false, a0.v, false, b0.v, (short)0, acc00, false, false);
    acc01 = __builtin_amdgcn_wmma_f32_16x16x32_bf16(false, a0.v, false, b1.v, (short)0, acc01, false, false);
    acc10 = __builtin_amdgcn_wmma_f32_16x16x32_bf16(false, a1.v, false, b0.v, (short)0, acc10, false, false);
    acc11 = __builtin_amdgcn_wmma_f32_16x16x32_bf16(false, a1.v, false, b1.v, (short)0, acc11, false, false);
  };

  // prologue: fill buffer 0 (B async, A via regs)
  {
    asyncB(0, 0, 0);
    uint4 a = loadA(0, 0);
    *(uint4*)&As[0][am * LDA + ag * 8] = a;
    asm volatile("s_wait_asynccnt 0x0" ::: "memory");
  }
  __syncthreads();

  // main loop: prefetch/async-fill next buffer overlapped with WMMA on current
  for (int kk = 0; kk < NK; ++kk) {
    uint4 an = {0, 0, 0, 0};
    const bool more = (kk + 1) < NK;
    if (more) {
      const int t  = (kk + 1) >> 3;
      const int c0 = ((kk + 1) & 7) * KB;
      asyncB((kk + 1) & 1, t, c0);              // async global->LDS for B
      an = loadA(t, c0);                        // registers for A
      if (kk + 2 < NK) {                        // L2 prefetch one step further
        const int t2  = (kk + 2) >> 3;
        const int c02 = ((kk + 2) & 7) * KB;
        __builtin_prefetch(bptr(t2, c02) + (size_t)bn_ * GC, 0, 1);
      }
    }
    compute(kk & 1);
    if (more) {
      *(uint4*)&As[(kk + 1) & 1][am * LDA + ag * 8] = an;
      asm volatile("s_wait_asynccnt 0x0" ::: "memory");
      __syncthreads();
    }
  }

  // epilogue: D frag lane holds 8 consecutive spatial rows at one out-channel
  const int nco   = lane & 15;
  const int rhalf = (lane >> 4) * 8;
  auto stfrag = [&](v8f acc, int fi, int fj) {
    const int col = n0 + wn + fj * 16 + nco;
    const int row = m0 + wm + fi * 16 + rhalf;
    float* op = out + ((size_t)(b * GO + col)) * GHW + row;
    union { v8f v; float4 f[2]; } u;
    u.v = acc;
    *(float4*)op       = u.f[0];
    *(float4*)(op + 4) = u.f[1];
  };
  stfrag(acc00, 0, 0);
  stfrag(acc01, 0, 1);
  stfrag(acc10, 1, 0);
  stfrag(acc11, 1, 1);
}

// =====================================================================
// launcher
// =====================================================================
extern "C" void kernel_launch(void* const* d_in, const int* in_sizes, int n_in,
                              void* d_out, int out_size, void* d_ws, size_t ws_size,
                              hipStream_t stream) {
  const float* x     = (const float*)d_in[0];
  const float* w1    = (const float*)d_in[1];
  const float* b1    = (const float*)d_in[2];
  const float* w2    = (const float*)d_in[3];
  const float* b2    = (const float*)d_in[4];
  const float* w_att = (const float*)d_in[5];
  const float* b_att = (const float*)d_in[6];
  float* out = (float*)d_out;

  char* ws = (char*)d_ws;
  size_t off = 0;
  auto take = [&](size_t bytes) {
    char* p = ws + off;
    off = (off + bytes + 255) & ~(size_t)255;
    return p;
  };
  __bf16* xbf  = (__bf16*)take((size_t)GB * GC * GHW * 2);          // 51.4 MB
  __bf16* wdyn = (__bf16*)take((size_t)GB * TAPS * GO * GC * 2);    // 37.7 MB
  float*  avgp = (float*)take((size_t)GB * GC * 25 * 4);
  float*  mxp  = (float*)take((size_t)GB * GC * 25 * 4);
  float*  relup= (float*)take((size_t)GB * GC * 9 * 4);

  cvt_kernel    <<<12544, 256, 0, stream>>>(x, xbf);
  pool_kernel   <<<GB * GC, 32, 0, stream>>>(x, avgp, mxp);
  dw_relu_kernel<<<(GB * GC) / 256, 256, 0, stream>>>(avgp, w1, b1, relup);
  wgen_kernel   <<<GB * GC, 256, 0, stream>>>(mxp, relup, w2, b2, w_att, b_att, wdyn);
  dynconv_wmma_kernel<<<dim3(GHW / BM, GO / BN, GB), 256, 0, stream>>>(xbf, wdyn, out);
}
